// GraphNeuralNet_79345225826944
// MI455X (gfx1250) — matrix operand order repaired and checked by reading, not compiled, not run on hardware
//
#include <hip/hip_runtime.h>
#include <stdint.h>

typedef __attribute__((ext_vector_type(16))) __bf16 v16bf;
typedef __attribute__((ext_vector_type(8)))  float  v8f;

union BF16Tile {
  v16bf v;
  uint32_t u[8];
  uint4 q[2];
};

__device__ __forceinline__ uint16_t f2bf(float f) {
  uint32_t a = __float_as_uint(f);
  return (uint16_t)((a + 0x7FFFu + ((a >> 16) & 1u)) >> 16);  // RNE
}

// ---------------- small prep kernels ----------------

__global__ void zero_f32_kernel(float* __restrict__ p, long long n) {
  long long i = (long long)blockIdx.x * blockDim.x + threadIdx.x;
  if (i < n) p[i] = 0.0f;
}

// elementwise f32 -> bf16 (node-feature table prep: halves gather bytes)
__global__ void f32_to_bf16_kernel(const float* __restrict__ in, uint16_t* __restrict__ out,
                                   long long n) {
  long long i = (long long)blockIdx.x * blockDim.x + threadIdx.x;
  if (i < n) out[i] = f2bf(in[i]);
}

// in [T][K][C] f32 -> out [T][C][K] bf16 (B-operand column is K-contiguous)
__global__ void transpose_w_kernel(const float* __restrict__ in, uint16_t* __restrict__ out,
                                   int T, int K, int C) {
  long long idx = (long long)blockIdx.x * blockDim.x + threadIdx.x;
  long long total = (long long)T * K * C;
  if (idx >= total) return;
  int c = (int)(idx % C);
  int k = (int)((idx / C) % K);
  int t = (int)(idx / ((long long)K * C));
  out[((long long)t * C + c) * K + k] = f2bf(in[idx]);
}

// per-(dst,type) in-edge counts; feature-independent, shared by both layers
__global__ void count_edges_kernel(const int* __restrict__ dst, const int* __restrict__ et,
                                   float* __restrict__ cnt, int E) {
  int e = blockIdx.x * blockDim.x + threadIdx.x;
  if (e < E) atomicAdd(&cnt[(long long)dst[e] * 3 + et[e]], 1.0f);
}

// mean over mailbox + convert to bf16: h[n][k] = acc[n][k] / max(cnt[n][k/32], 1)
__global__ void node_prep_kernel(const float* __restrict__ acc, const float* __restrict__ cnt,
                                 uint16_t* __restrict__ hbf, int N) {
  long long idx = (long long)blockIdx.x * blockDim.x + threadIdx.x;
  if (idx >= (long long)N * 96) return;
  int n = (int)(idx / 96), k = (int)(idx % 96);
  float cc = cnt[(long long)n * 3 + (k >> 5)];
  cc = cc < 1.0f ? 1.0f : cc;
  hbf[idx] = f2bf(acc[idx] / cc);
}

// ---------------- edge message MLP (WMMA) ----------------
// 1 wave = 16 edges. A operand gathered from the L2-resident bf16 node table
// straight into the ISA 16-bit A layout; B operand served from LDS-staged
// transposed weights (all 6 tiles batch-loaded per K-block so the ds clause
// drains while WMMAs run); 6 accumulators = 3 types x 2 col-tiles;
// bias+relu+masked atomic mean-scatter epilogue.
template<int DIM>   // per-node feature dim (128 for layer 0, 64 for layer 1)
__global__ void edge_mlp_kernel(const uint16_t* __restrict__ nfb,  // [N][DIM] bf16
                                const int* __restrict__ src, const int* __restrict__ dst,
                                const int* __restrict__ etype,
                                const uint16_t* __restrict__ Wt,   // [3][32][2*DIM] bf16
                                const float* __restrict__ bias,    // [3][32]
                                float* __restrict__ acc,           // [N][96] f32
                                int E) {
  constexpr int KBLK = (2 * DIM) / 32;
  constexpr int HALF = KBLK / 2;
  constexpr int WELEMS = 3 * 32 * 2 * DIM;            // u16 elements of weight table

  __shared__ __align__(16) uint16_t sW[WELEMS];       // 49 KB (DIM=128) / 24.5 KB (DIM=64)
  for (int i = threadIdx.x; i < WELEMS / 8; i += blockDim.x)
    ((uint4*)sW)[i] = ((const uint4*)Wt)[i];
  __syncthreads();

  const int lane = threadIdx.x & 31;
  const int tile = blockIdx.x * (blockDim.x >> 5) + (threadIdx.x >> 5);
  if (tile * 16 >= E) return;          // wave-uniform (after barrier): EXEC all-1s
  const int sel = lane >> 4;           // half-lane group
  const int row = lane & 15;           // A row (edge-in-tile) / B column
  int eg = tile * 16 + row; if (eg > E - 1) eg = E - 1;
  const int s_node = src[eg];
  const int d_node = dst[eg];

  const v8f vzero = {0, 0, 0, 0, 0, 0, 0, 0};
  v8f c[6];
#pragma unroll
  for (int i = 0; i < 6; ++i) c[i] = vzero;

  for (int kb = 0; kb < KBLK; ++kb) {
    // K blocks 0..HALF-1 come from nf[src], rest from nf[dst] (concat)
    const int node = (kb < HALF) ? s_node : d_node;
    const int kloc = ((kb < HALF) ? kb : kb - HALF) * 32;
    // lane<16: K {kloc+0..7, kloc+16..23}; lane>=16: K {kloc+8..15, kloc+24..31}
    BF16Tile A;
    const uint16_t* p = nfb + (size_t)node * DIM + kloc + sel * 8;
    A.q[0] = *(const uint4*)p;
    A.q[1] = *(const uint4*)(p + 16);

    // batch-load all 6 B tiles (one ds clause), then issue the 6 WMMAs
    BF16Tile B[6];
#pragma unroll
    for (int i = 0; i < 6; ++i) {
      const uint16_t* w = sW + (i * 16 + row) * (2 * DIM) + kb * 32 + sel * 8;
      B[i].q[0] = *(const uint4*)w;
      B[i].q[1] = *(const uint4*)(w + 16);
    }
#pragma unroll
    for (int i = 0; i < 6; ++i)
      c[i] = __builtin_amdgcn_wmma_f32_16x16x32_bf16(
          false, A.v, false, B[i].v, (short)0, c[i], false, false);
  }

  // epilogue: rows held by this lane are r + 8*sel, column is ct*16+row
  int et8[8], dn8[8];
#pragma unroll
  for (int r = 0; r < 8; ++r) {
    int e = tile * 16 + r + 8 * sel;
    int ec = e > E - 1 ? E - 1 : e;
    et8[r] = etype[ec];
    dn8[r] = dst[ec];
  }
#pragma unroll
  for (int t = 0; t < 3; ++t) {
#pragma unroll
    for (int ct = 0; ct < 2; ++ct) {
      const float b = bias[t * 32 + ct * 16 + row];
#pragma unroll
      for (int r = 0; r < 8; ++r) {
        int e = tile * 16 + r + 8 * sel;
        if (e < E && et8[r] == t) {
          float v = c[t * 2 + ct][r] + b;
          v = v > 0.0f ? v : 0.0f;
          atomicAdd(&acc[(size_t)dn8[r] * 96 + t * 32 + ct * 16 + row], v);
        }
      }
    }
  }
}

// ---------------- node update MLP (WMMA) ----------------
// 1 wave = 16 nodes; K = 96 (3 blocks); computes both node-type weight sets
// (LDS-staged, B tiles batch-loaded) and predicates the store on node_type.
template<int OUTDIM, bool OUTBF>
__global__ void node_mlp_kernel(const uint16_t* __restrict__ hbf,  // [N][96] bf16
                                const uint16_t* __restrict__ Wt,   // [2][OUTDIM][96] bf16
                                const float* __restrict__ bias,    // [2][OUTDIM]
                                const int* __restrict__ ntype,
                                float* __restrict__ outf, uint16_t* __restrict__ outb,
                                int N) {
  constexpr int WELEMS = 2 * OUTDIM * 96;
  __shared__ __align__(16) uint16_t sW[WELEMS];       // 24.5 KB / 49 KB
  for (int i = threadIdx.x; i < WELEMS / 8; i += blockDim.x)
    ((uint4*)sW)[i] = ((const uint4*)Wt)[i];
  __syncthreads();

  const int lane = threadIdx.x & 31;
  const int tile = blockIdx.x * (blockDim.x >> 5) + (threadIdx.x >> 5);
  if (tile * 16 >= N) return;          // wave-uniform (after barrier)
  const int sel = lane >> 4;
  const int row = lane & 15;
  int arow = tile * 16 + row; if (arow > N - 1) arow = N - 1;

  BF16Tile A[3];
#pragma unroll
  for (int kb = 0; kb < 3; ++kb) {
    const uint16_t* p = hbf + (size_t)arow * 96 + kb * 32 + sel * 8;
    A[kb].q[0] = *(const uint4*)p;
    A[kb].q[1] = *(const uint4*)(p + 16);
  }

  int nt8[8];
#pragma unroll
  for (int r = 0; r < 8; ++r) {
    int n = tile * 16 + r + 8 * sel;
    int nc = n > N - 1 ? N - 1 : n;
    nt8[r] = ntype[nc];
  }

  const v8f vzero = {0, 0, 0, 0, 0, 0, 0, 0};
  for (int nt = 0; nt < 2; ++nt) {
    for (int ct = 0; ct < OUTDIM / 16; ++ct) {
      // batch-load the 3 B tiles, then the 3 chained WMMAs
      BF16Tile B[3];
#pragma unroll
      for (int kb = 0; kb < 3; ++kb) {
        const uint16_t* w = sW + (nt * OUTDIM + ct * 16 + row) * 96 + kb * 32 + sel * 8;
        B[kb].q[0] = *(const uint4*)w;
        B[kb].q[1] = *(const uint4*)(w + 16);
      }
      v8f c = vzero;
#pragma unroll
      for (int kb = 0; kb < 3; ++kb)
        c = __builtin_amdgcn_wmma_f32_16x16x32_bf16(
            false, A[kb].v, false, B[kb].v, (short)0, c, false, false);

      const float b = bias[nt * OUTDIM + ct * 16 + row];
#pragma unroll
      for (int r = 0; r < 8; ++r) {
        int n = tile * 16 + r + 8 * sel;
        if (n < N && nt8[r] == nt) {
          float v = c[r] + b;
          v = v > 0.0f ? v : 0.0f;
          if (OUTBF) outb[(size_t)n * OUTDIM + ct * 16 + row] = f2bf(v);
          else       outf[(size_t)n * OUTDIM + ct * 16 + row] = v;
        }
      }
    }
  }
}

// ---------------- launch ----------------

extern "C" void kernel_launch(void* const* d_in, const int* in_sizes, int n_in,
                              void* d_out, int out_size, void* d_ws, size_t ws_size,
                              hipStream_t stream) {
  const float* nf   = (const float*)d_in[0];
  const int*   eidx = (const int*)d_in[1];
  const int*   etyp = (const int*)d_in[2];
  const int*   ntyp = (const int*)d_in[3];
  const float* eW0  = (const float*)d_in[4];
  const float* eb0  = (const float*)d_in[5];
  const float* nW0  = (const float*)d_in[6];
  const float* nb0  = (const float*)d_in[7];
  const float* eW1  = (const float*)d_in[8];
  const float* eb1  = (const float*)d_in[9];
  const float* nW1  = (const float*)d_in[10];
  const float* nb1  = (const float*)d_in[11];

  const int E = in_sizes[2];   // edge_type has E elements
  const int N = in_sizes[3];   // node_type has N elements
  const int* src = eidx;
  const int* dst = eidx + E;

  // workspace layout (all chunks 16B-aligned for the given N)
  float*    acc  = (float*)d_ws;                       // [N][96]
  float*    cnt  = acc + (size_t)N * 96;               // [N][3]
  uint16_t* hbf  = (uint16_t*)(cnt + (size_t)N * 3);   // [N][96]  bf16 mailbox means
  uint16_t* h1bf = hbf + (size_t)N * 96;               // [N][64]  bf16 layer-0 output
  uint16_t* nfb  = h1bf + (size_t)N * 64;              // [N][128] bf16 node features
  uint16_t* eW0t = nfb + (size_t)N * 128;              // [3][32][256]
  uint16_t* eW1t = eW0t + 3 * 32 * 256;                // [3][32][128]
  uint16_t* nW0t = eW1t + 3 * 32 * 128;                // [2][64][96]
  uint16_t* nW1t = nW0t + 2 * 64 * 96;                 // [2][128][96]

  auto g1 = [](long long n) { return (int)((n + 255) / 256); };

  // prep: bf16 node table + transposed bf16 weights
  f32_to_bf16_kernel<<<g1((long long)N * 128), 256, 0, stream>>>(nf, nfb, (long long)N * 128);
  transpose_w_kernel<<<g1(3LL * 256 * 32), 256, 0, stream>>>(eW0, eW0t, 3, 256, 32);
  transpose_w_kernel<<<g1(3LL * 128 * 32), 256, 0, stream>>>(eW1, eW1t, 3, 128, 32);
  transpose_w_kernel<<<g1(2LL * 96 * 64),  256, 0, stream>>>(nW0, nW0t, 2, 96, 64);
  transpose_w_kernel<<<g1(2LL * 96 * 128), 256, 0, stream>>>(nW1, nW1t, 2, 96, 128);

  // zero acc+cnt (contiguous) and build per-(dst,type) counts once
  zero_f32_kernel<<<g1((long long)N * 99), 256, 0, stream>>>(acc, (long long)N * 99);
  count_edges_kernel<<<g1(E), 256, 0, stream>>>(dst, etyp, cnt, E);

  const int eblocks = ((E + 15) / 16 + 7) / 8;   // 8 waves/block, 16 edges/wave
  const int nblocks = ((N + 15) / 16 + 7) / 8;

  // layer 0: 128 -> (edge MLP 32x3) -> 64
  edge_mlp_kernel<128><<<eblocks, 256, 0, stream>>>(nfb, src, dst, etyp, eW0t, eb0, acc, E);
  node_prep_kernel<<<g1((long long)N * 96), 256, 0, stream>>>(acc, cnt, hbf, N);
  node_mlp_kernel<64, true><<<nblocks, 256, 0, stream>>>(hbf, nW0t, nb0, ntyp, nullptr, h1bf, N);

  // layer 1: 64 -> (edge MLP 32x3) -> 128
  zero_f32_kernel<<<g1((long long)N * 96), 256, 0, stream>>>(acc, (long long)N * 96);
  edge_mlp_kernel<64><<<eblocks, 256, 0, stream>>>(h1bf, src, dst, etyp, eW1t, eb1, acc, E);
  node_prep_kernel<<<g1((long long)N * 96), 256, 0, stream>>>(acc, cnt, hbf, N);
  node_mlp_kernel<128, false><<<nblocks, 256, 0, stream>>>(hbf, nW1t, nb1, ntyp, (float*)d_out, nullptr, N);
}
